// multihead_layer_5231270166596
// MI455X (gfx1250) — matrix operand order repaired and checked by reading, not compiled, not run on hardware
//
#include <hip/hip_runtime.h>

// MI455X / gfx1250: wave32, WMMA 16x16x32 f16 -> f32, TDM tensor_load_to_lds staging.
// Pipeline:
//   prep:       transpose weights to f16 "Bt" fragment layout (B[n][k] contiguous in k)
//   init:       zero se/agg, set segment-max keys to ord(-inf)
//   proj_nodes: X[N,128] x {Wq,Wk,Wv} -> q(f32), k(f16), v(f16), fused per-head LN
//               (activation tile staged to LDS by the Tensor Data Mover)
//   proj_edges: Xe[E,128] x {Wke,Wve} -> ke(f16), ve(f16), fused per-head LN
//   relk_qk:    per (16-edge tile, head): A=[ke | k[src]] (16x32) WMMA relk_W, LN,
//               dot with q[dst] -> qk[E,H]; atomicMax segment max (ordered-uint)
//   lse_sum:    atomicAdd exp(qk - m) per (dst,h)
//   lse_fin:    lse = log(se) + m
//   relv_agg:   A=[ve | v[src]] WMMA relv_W, LN; (qk - lse[dst]) * rel_v
//               scatter atomicAdd into agg[N,H,16]
//   zout:       agg[16x16] WMMA z_W (K zero-padded to 32), +b, relu, LN -> out[N,128]

typedef __attribute__((ext_vector_type(16))) _Float16 v16h;
typedef __attribute__((ext_vector_type(8)))  float    v8f;
typedef __attribute__((ext_vector_type(4)))  unsigned v4u;
typedef __attribute__((ext_vector_type(8)))  unsigned v8u;

union Frag16 {
  v16h v;
  uint4 q[2];
  _Float16 h[16];
};

__device__ __forceinline__ float lane_sum16(float x) {
  #pragma unroll
  for (int off = 1; off < 16; off <<= 1) x += __shfl_xor(x, off, 16);
  return x;
}

// order-preserving float <-> uint for atomicMax-based segment max
__device__ __forceinline__ unsigned f2ord(float f) {
  unsigned u = __float_as_uint(f);
  return (u & 0x80000000u) ? ~u : (u | 0x80000000u);
}
__device__ __forceinline__ float ord2f(unsigned u) {
  unsigned v = (u & 0x80000000u) ? (u & 0x7fffffffu) : ~u;
  return __uint_as_float(v);
}

#define WMMA_F16(A, B, C) \
  __builtin_amdgcn_wmma_f32_16x16x32_f16(false, (A), false, (B), (short)0, (C), false, false)

// Tensor Data Mover: DMA a [tileRows x 128] f32 tile (row stride 128) from global
// into LDS. rowsRemaining = tensor rows from gsrc; rows past it read as zero.
// D# per CDNA5 ISA 8.3/8.4: group0 = {count, lds_addr, global_addr, type=2},
// group1 = {data_size=4B, tensor_dim0=128, tensor_dim1, tile 128x128, stride 128}.
__device__ __forceinline__ void tdm_load_tile_f32(const float* gsrc, unsigned ldsOff,
                                                  int rowsRemaining) {
  unsigned long long ga = (unsigned long long)(__SIZE_TYPE__)gsrc;
  unsigned dim1 = (unsigned)rowsRemaining;
  v4u g0;
  g0[0] = 1u;                                                  // count=1, user mode
  g0[1] = ldsOff;                                              // lds_addr (bytes)
  g0[2] = (unsigned)(ga & 0xffffffffu);                        // global_addr[31:0]
  g0[3] = (unsigned)((ga >> 32) & 0x01ffffffu) | 0x80000000u;  // addr[56:32] | type=2
  v8u g1;
  g1[0] = 2u << 16;                        // data_size = 2 (4 bytes)
  g1[1] = 128u << 16;                      // tensor_dim0 lo16 = 128
  g1[2] = (dim1 & 0xffffu) << 16;          // tensor_dim1 lo16 | tensor_dim0 hi16(0)
  g1[3] = (128u << 16) | (dim1 >> 16);     // tile_dim0 = 128 | tensor_dim1 hi16
  g1[4] = 128u;                            // tile_dim1 = 128, tile_dim2 = 0
  g1[5] = 128u;                            // tensor_dim0_stride lo32 = 128
  g1[6] = 0u;                              // dim0_stride hi | dim1_stride lo16
  g1[7] = 0u;
  asm volatile("tensor_load_to_lds %0, %1" : : "s"(g0), "s"(g1) : "memory");
  __builtin_amdgcn_s_wait_tensorcnt(0);
}

__device__ __forceinline__ void cvt8(_Float16* d, float4 a, float4 b) {
  d[0] = (_Float16)a.x; d[1] = (_Float16)a.y; d[2] = (_Float16)a.z; d[3] = (_Float16)a.w;
  d[4] = (_Float16)b.x; d[5] = (_Float16)b.y; d[6] = (_Float16)b.z; d[7] = (_Float16)b.w;
}

// ---------------------------------------------------------------- prep weights
__global__ void prep_kernel(const float* __restrict__ Wq, const float* __restrict__ Wk,
                            const float* __restrict__ Wv, const float* __restrict__ Wke,
                            const float* __restrict__ Wve, const float* __restrict__ relkW,
                            const float* __restrict__ relvW, const float* __restrict__ zW,
                            _Float16* __restrict__ wTnode, _Float16* __restrict__ wTedge,
                            _Float16* __restrict__ relkT, _Float16* __restrict__ relvT,
                            _Float16* __restrict__ zwT) {
  const int WN = 128 * 128;
  int stride = gridDim.x * blockDim.x;
  int tid = blockIdx.x * blockDim.x + threadIdx.x;
  // node weights: Bt[(h*16+d)][f] = W[h][f][d]
  for (int i = tid; i < 3 * WN; i += stride) {
    int sel = i / WN, j = i - sel * WN;
    int n = j >> 7, f = j & 127;
    int hh = n >> 4, d = n & 15;
    const float* W = (sel == 0) ? Wq : (sel == 1) ? Wk : Wv;
    wTnode[i] = (_Float16)W[(hh * 128 + f) * 16 + d];
  }
  for (int i = tid; i < 2 * WN; i += stride) {
    int sel = i / WN, j = i - sel * WN;
    int n = j >> 7, f = j & 127;
    int hh = n >> 4, d = n & 15;
    const float* W = (sel == 0) ? Wke : Wve;
    wTedge[i] = (_Float16)W[(hh * 128 + f) * 16 + d];
  }
  // rel MLPs: [32,16] -> Bt[d][kk];  z: [16,16] -> Bt[d][kk] zero-padded to K=32
  for (int i = tid; i < 16 * 32; i += stride) {
    int d = i >> 5, kk = i & 31;
    relkT[i] = (_Float16)relkW[kk * 16 + d];
    relvT[i] = (_Float16)relvW[kk * 16 + d];
    zwT[i] = (kk < 16) ? (_Float16)zW[kk * 16 + d] : (_Float16)0.0f;
  }
}

__global__ void init_kernel(unsigned* __restrict__ mkey, float* __restrict__ se,
                            float* __restrict__ agg, int NH, int NF) {
  int stride = gridDim.x * blockDim.x;
  unsigned ninf = f2ord(-__builtin_inff());
  for (int i = blockIdx.x * blockDim.x + threadIdx.x; i < NF; i += stride) agg[i] = 0.0f;
  for (int i = blockIdx.x * blockDim.x + threadIdx.x; i < NH; i += stride) {
    mkey[i] = ninf;
    se[i] = 0.0f;
  }
}

// ---------------------------------------------------------------- projections
// 128 rows per block staged to LDS by TDM; 8 waves, wave w does rows [16w,16w+16);
// fused per-head LayerNorm in the WMMA epilogue.
__global__ void __launch_bounds__(256) proj_nodes_kernel(
    const float* __restrict__ x, int N, const _Float16* __restrict__ wT,
    const float* __restrict__ gq, const float* __restrict__ bq,
    const float* __restrict__ gk, const float* __restrict__ bk,
    const float* __restrict__ gv, const float* __restrict__ bv,
    float* __restrict__ qOut, _Float16* __restrict__ kOut, _Float16* __restrict__ vOut) {
  __shared__ __align__(16) float xsf[128 * 128];
  int rowBase = blockIdx.x * 128;
  // warm the WGP cache with the (shared) weight tables while the DMA runs
  for (int i = threadIdx.x; i < 3 * 128; i += 256) __builtin_prefetch(wT + i * 128, 0, 0);
  if (threadIdx.x < 32) {
    tdm_load_tile_f32(x + (size_t)rowBase * 128, (unsigned)(__SIZE_TYPE__)xsf, N - rowBase);
  }
  __syncthreads();
  int w = threadIdx.x >> 5, lane = threadIdx.x & 31;
  int m = lane & 15, hi = lane >> 4;
  Frag16 a[4];
  #pragma unroll
  for (int kt = 0; kt < 4; kt++) {  // A 16x32: K chunks at kbase and kbase+16
    const float* p = &xsf[(w * 16 + m) * 128 + kt * 32 + hi * 8];
    cvt8(&a[kt].h[0], *(const float4*)p, *(const float4*)(p + 4));
    cvt8(&a[kt].h[8], *(const float4*)(p + 16), *(const float4*)(p + 20));
  }
  #pragma unroll
  for (int sel = 0; sel < 3; sel++) {
    const _Float16* W = wT + sel * 128 * 128;
    const float* g = (sel == 0) ? gq : (sel == 1) ? gk : gv;
    const float* bb = (sel == 0) ? bq : (sel == 1) ? bk : bv;
    for (int hh = 0; hh < 8; hh++) {
      v8f acc = {};
      #pragma unroll
      for (int kt = 0; kt < 4; kt++) {
        Frag16 bf;
        const _Float16* p = W + (hh * 16 + m) * 128 + kt * 32 + hi * 8;
        bf.q[0] = *(const uint4*)p;
        bf.q[1] = *(const uint4*)(p + 16);
        acc = WMMA_F16(a[kt].v, bf.v, acc);
      }
      float gs = g[hh * 16 + m], bs = bb[hh * 16 + m];
      #pragma unroll
      for (int r = 0; r < 8; r++) {
        float val = acc[r];
        float s = lane_sum16(val);
        float s2 = lane_sum16(val * val);
        float mean = s * (1.0f / 16.0f);
        float var = s2 * (1.0f / 16.0f) - mean * mean;
        float out = (val - mean) * rsqrtf(var + 1e-5f) * gs + bs;
        int grow = rowBase + w * 16 + r + hi * 8;
        if (grow < N) {
          int col = hh * 16 + m;
          if (sel == 0)      qOut[grow * 128 + col] = out;
          else if (sel == 1) kOut[grow * 128 + col] = (_Float16)out;
          else               vOut[grow * 128 + col] = (_Float16)out;
        }
      }
    }
  }
}

__global__ void __launch_bounds__(256) proj_edges_kernel(
    const float* __restrict__ x, int E, const _Float16* __restrict__ wT,
    const float* __restrict__ gke, const float* __restrict__ bke,
    const float* __restrict__ gve, const float* __restrict__ bve,
    _Float16* __restrict__ keOut, _Float16* __restrict__ veOut) {
  __shared__ __align__(16) float xsf[128 * 128];
  int rowBase = blockIdx.x * 128;
  for (int i = threadIdx.x; i < 2 * 128; i += 256) __builtin_prefetch(wT + i * 128, 0, 0);
  if (threadIdx.x < 32) {
    tdm_load_tile_f32(x + (size_t)rowBase * 128, (unsigned)(__SIZE_TYPE__)xsf, E - rowBase);
  }
  __syncthreads();
  int w = threadIdx.x >> 5, lane = threadIdx.x & 31;
  int m = lane & 15, hi = lane >> 4;
  Frag16 a[4];
  #pragma unroll
  for (int kt = 0; kt < 4; kt++) {
    const float* p = &xsf[(w * 16 + m) * 128 + kt * 32 + hi * 8];
    cvt8(&a[kt].h[0], *(const float4*)p, *(const float4*)(p + 4));
    cvt8(&a[kt].h[8], *(const float4*)(p + 16), *(const float4*)(p + 20));
  }
  #pragma unroll
  for (int sel = 0; sel < 2; sel++) {
    const _Float16* W = wT + sel * 128 * 128;
    const float* g = (sel == 0) ? gke : gve;
    const float* bb = (sel == 0) ? bke : bve;
    for (int hh = 0; hh < 8; hh++) {
      v8f acc = {};
      #pragma unroll
      for (int kt = 0; kt < 4; kt++) {
        Frag16 bf;
        const _Float16* p = W + (hh * 16 + m) * 128 + kt * 32 + hi * 8;
        bf.q[0] = *(const uint4*)p;
        bf.q[1] = *(const uint4*)(p + 16);
        acc = WMMA_F16(a[kt].v, bf.v, acc);
      }
      float gs = g[hh * 16 + m], bs = bb[hh * 16 + m];
      #pragma unroll
      for (int r = 0; r < 8; r++) {
        float val = acc[r];
        float s = lane_sum16(val);
        float s2 = lane_sum16(val * val);
        float mean = s * (1.0f / 16.0f);
        float var = s2 * (1.0f / 16.0f) - mean * mean;
        float out = (val - mean) * rsqrtf(var + 1e-5f) * gs + bs;
        int grow = rowBase + w * 16 + r + hi * 8;
        if (grow < E) {
          int col = hh * 16 + m;
          if (sel == 0) keOut[grow * 128 + col] = (_Float16)out;
          else          veOut[grow * 128 + col] = (_Float16)out;
        }
      }
    }
  }
}

// --------------------------------------------------- rel_k + qk + segment max
// One block = one 16-edge tile; wave w handles head h=w. A = [ke | k[src]] 16x32.
__global__ void __launch_bounds__(256) relk_qk_kernel(
    const _Float16* __restrict__ keH, const _Float16* __restrict__ kH,
    const float* __restrict__ qF, const int* __restrict__ srcI, const int* __restrict__ dstI,
    const _Float16* __restrict__ relkT, const float* __restrict__ rb,
    const float* __restrict__ rg, const float* __restrict__ rbeta,
    float* __restrict__ qkOut, unsigned* __restrict__ mkey, int E) {
  int hh = threadIdx.x >> 5, lane = threadIdx.x & 31;
  int m = lane & 15, hi = lane >> 4;
  int eBase = blockIdx.x * 16;
  int eA = eBase + m;
  if (eA >= E) eA = E - 1;
  int sA = srcI[eA];
  Frag16 a, b;
  a.q[0] = *(const uint4*)(keH + eA * 128 + hh * 16 + hi * 8);  // K 0..15 = ke
  a.q[1] = *(const uint4*)(kH + sA * 128 + hh * 16 + hi * 8);   // K 16..31 = k[src]
  const _Float16* p = relkT + m * 32 + hi * 8;
  b.q[0] = *(const uint4*)p;
  b.q[1] = *(const uint4*)(p + 16);
  v8f acc = {};
  acc = WMMA_F16(a.v, b.v, acc);
  float bias = rb[m], g = rg[m], beta = rbeta[m];
  float qk_r[8];
  int dst_r[8];
  #pragma unroll
  for (int r = 0; r < 8; r++) {
    float val = acc[r] + bias;
    float s = lane_sum16(val);
    float s2 = lane_sum16(val * val);
    float mean = s * (1.0f / 16.0f);
    float var = s2 * (1.0f / 16.0f) - mean * mean;
    float rk = (val - mean) * rsqrtf(var + 1e-5f) * g + beta;
    int er = eBase + r + hi * 8;
    if (er >= E) er = E - 1;
    int de = dstI[er];
    dst_r[r] = de;
    float qv = qF[de * 128 + hh * 16 + m];
    qk_r[r] = lane_sum16(rk * qv) * 0.25f;
  }
  if (m == 0) {
    #pragma unroll
    for (int r = 0; r < 8; r++) {
      int er = eBase + r + hi * 8;
      if (er < E) {
        qkOut[er * 8 + hh] = qk_r[r];
        atomicMax(&mkey[dst_r[r] * 8 + hh], f2ord(qk_r[r]));
      }
    }
  }
}

__global__ void lse_sum_kernel(const float* __restrict__ qk, const int* __restrict__ dstI,
                               const unsigned* __restrict__ mkey, float* __restrict__ se, int E) {
  int i = blockIdx.x * blockDim.x + threadIdx.x;
  if (i >= E * 8) return;
  int e = i >> 3, hh = i & 7;
  int de = dstI[e];
  float mv = ord2f(mkey[de * 8 + hh]);
  atomicAdd(&se[de * 8 + hh], expf(qk[i] - mv));
}

__global__ void lse_fin_kernel(const float* __restrict__ se, const unsigned* __restrict__ mkey,
                               float* __restrict__ lse, int NH) {
  int i = blockIdx.x * blockDim.x + threadIdx.x;
  if (i < NH) lse[i] = logf(se[i]) + ord2f(mkey[i]);
}

// ------------------------------------------------------- rel_v + scatter aggregation
__global__ void __launch_bounds__(256) relv_agg_kernel(
    const _Float16* __restrict__ veH, const _Float16* __restrict__ vH,
    const int* __restrict__ srcI, const int* __restrict__ dstI,
    const _Float16* __restrict__ relvT, const float* __restrict__ rb,
    const float* __restrict__ rg, const float* __restrict__ rbeta,
    const float* __restrict__ qk, const float* __restrict__ lse,
    float* __restrict__ agg, int E) {
  int hh = threadIdx.x >> 5, lane = threadIdx.x & 31;
  int m = lane & 15, hi = lane >> 4;
  int eBase = blockIdx.x * 16;
  int eA = eBase + m;
  if (eA >= E) eA = E - 1;
  int sA = srcI[eA];
  Frag16 a, b;
  a.q[0] = *(const uint4*)(veH + eA * 128 + hh * 16 + hi * 8);
  a.q[1] = *(const uint4*)(vH + sA * 128 + hh * 16 + hi * 8);
  const _Float16* p = relvT + m * 32 + hi * 8;
  b.q[0] = *(const uint4*)p;
  b.q[1] = *(const uint4*)(p + 16);
  v8f acc = {};
  acc = WMMA_F16(a.v, b.v, acc);
  float bias = rb[m], g = rg[m], beta = rbeta[m];
  #pragma unroll
  for (int r = 0; r < 8; r++) {
    float val = acc[r] + bias;
    float s = lane_sum16(val);
    float s2 = lane_sum16(val * val);
    float mean = s * (1.0f / 16.0f);
    float var = s2 * (1.0f / 16.0f) - mean * mean;
    float rv = (val - mean) * rsqrtf(var + 1e-5f) * g + beta;
    int er = eBase + r + hi * 8;
    bool ok = (er < E);
    if (!ok) er = E - 1;
    int de = dstI[er];
    float logw = qk[er * 8 + hh] - lse[de * 8 + hh];
    if (ok) atomicAdd(&agg[de * 128 + hh * 16 + m], logw * rv);
  }
}

// ------------------------------------------------------------ z-fc + relu + LN
__global__ void __launch_bounds__(256) zout_kernel(
    const float* __restrict__ agg, const _Float16* __restrict__ zwT,
    const float* __restrict__ zb, const float* __restrict__ zg,
    const float* __restrict__ zbeta, float* __restrict__ out, int N) {
  int hh = threadIdx.x >> 5, lane = threadIdx.x & 31;
  int m = lane & 15, hi = lane >> 4;
  int nodeBase = blockIdx.x * 16;
  int nA = nodeBase + m;
  if (nA >= N) nA = N - 1;
  Frag16 a, b;
  const float* ap = agg + nA * 128 + hh * 16 + hi * 8;
  #pragma unroll
  for (int i = 0; i < 8; i++) {
    a.h[i] = (_Float16)ap[i];
    a.h[8 + i] = (_Float16)0.0f;  // zero-pad K 16..31
  }
  const _Float16* p = zwT + m * 32 + hi * 8;
  b.q[0] = *(const uint4*)p;
  b.q[1] = *(const uint4*)(p + 16);
  v8f acc = {};
  acc = WMMA_F16(a.v, b.v, acc);
  float bias = zb[m], g = zg[m], beta = zbeta[m];
  #pragma unroll
  for (int r = 0; r < 8; r++) {
    float val = fmaxf(acc[r] + bias, 0.0f);
    float s = lane_sum16(val);
    float s2 = lane_sum16(val * val);
    float mean = s * (1.0f / 16.0f);
    float var = s2 * (1.0f / 16.0f) - mean * mean;
    float o = (val - mean) * rsqrtf(var + 1e-5f) * g + beta;
    int gn = nodeBase + r + hi * 8;
    if (gn < N) out[gn * 128 + hh * 16 + m] = o;
  }
}

// ---------------------------------------------------------------------- launch
extern "C" void kernel_launch(void* const* d_in, const int* in_sizes, int n_in,
                              void* d_out, int out_size, void* d_ws, size_t ws_size,
                              hipStream_t stream) {
  (void)n_in; (void)out_size; (void)ws_size;
  const float* x_node = (const float*)d_in[0];
  const float* x_edge = (const float*)d_in[1];
  const int* srcI = (const int*)d_in[2];
  const int* dstI = (const int*)d_in[3];
  const float* Wq = (const float*)d_in[4];
  const float* Wk = (const float*)d_in[5];
  const float* Wv = (const float*)d_in[6];
  const float* Wke = (const float*)d_in[7];
  const float* Wve = (const float*)d_in[8];
  const float* gq = (const float*)d_in[9];   const float* bq = (const float*)d_in[10];
  const float* gk = (const float*)d_in[11];  const float* bk = (const float*)d_in[12];
  const float* gv = (const float*)d_in[13];  const float* bv = (const float*)d_in[14];
  const float* gke = (const float*)d_in[15]; const float* bke = (const float*)d_in[16];
  const float* gve = (const float*)d_in[17]; const float* bve = (const float*)d_in[18];
  const float* relkW = (const float*)d_in[19]; const float* relkB = (const float*)d_in[20];
  const float* relkG = (const float*)d_in[21]; const float* relkBe = (const float*)d_in[22];
  const float* relvW = (const float*)d_in[23]; const float* relvB = (const float*)d_in[24];
  const float* relvG = (const float*)d_in[25]; const float* relvBe = (const float*)d_in[26];
  const float* zW = (const float*)d_in[27]; const float* zB = (const float*)d_in[28];
  const float* zG = (const float*)d_in[29]; const float* zBe = (const float*)d_in[30];
  float* outP = (float*)d_out;

  const int N = in_sizes[0] / 128;
  const int E = in_sizes[1] / 128;

  // workspace carving (256B aligned)
  char* base = (char*)d_ws;
  size_t off = 0;
  auto carve = [&](size_t bytes) -> char* {
    char* pp = base + off;
    off = (off + bytes + 255) & ~(size_t)255;
    return pp;
  };
  _Float16* wTnode = (_Float16*)carve((size_t)3 * 128 * 128 * 2);
  _Float16* wTedge = (_Float16*)carve((size_t)2 * 128 * 128 * 2);
  _Float16* relkT = (_Float16*)carve(16 * 32 * 2);
  _Float16* relvT = (_Float16*)carve(16 * 32 * 2);
  _Float16* zwT = (_Float16*)carve(16 * 32 * 2);
  float* qF = (float*)carve((size_t)N * 128 * 4);
  _Float16* kH = (_Float16*)carve((size_t)N * 128 * 2);
  _Float16* vH = (_Float16*)carve((size_t)N * 128 * 2);
  _Float16* keH = (_Float16*)carve((size_t)E * 128 * 2);
  _Float16* veH = (_Float16*)carve((size_t)E * 128 * 2);
  float* qkP = (float*)carve((size_t)E * 8 * 4);
  unsigned* mkey = (unsigned*)carve((size_t)N * 8 * 4);
  float* seP = (float*)carve((size_t)N * 8 * 4);
  float* lseP = (float*)carve((size_t)N * 8 * 4);
  float* aggP = (float*)carve((size_t)N * 128 * 4);

  prep_kernel<<<256, 256, 0, stream>>>(Wq, Wk, Wv, Wke, Wve, relkW, relvW, zW,
                                       wTnode, wTedge, relkT, relvT, zwT);
  init_kernel<<<(N * 128 + 255) / 256, 256, 0, stream>>>(mkey, seP, aggP, N * 8, N * 128);
  proj_nodes_kernel<<<(N + 127) / 128, 256, 0, stream>>>(
      x_node, N, wTnode, gq, bq, gk, bk, gv, bv, qF, kH, vH);
  proj_edges_kernel<<<(E + 127) / 128, 256, 0, stream>>>(
      x_edge, E, wTedge, gke, bke, gve, bve, keH, veH);
  relk_qk_kernel<<<(E + 15) / 16, 256, 0, stream>>>(
      keH, kH, qF, srcI, dstI, relkT, relkB, relkG, relkBe, qkP, mkey, E);
  lse_sum_kernel<<<(E * 8 + 255) / 256, 256, 0, stream>>>(qkP, dstI, mkey, seP, E);
  lse_fin_kernel<<<(N * 8 + 255) / 256, 256, 0, stream>>>(seP, mkey, lseP, N * 8);
  relv_agg_kernel<<<(E + 15) / 16, 256, 0, stream>>>(
      veH, vH, srcI, dstI, relvT, relvB, relvG, relvBe, qkP, lseP, aggP, E);
  zout_kernel<<<(N + 15) / 16, 256, 0, stream>>>(aggP, zwT, zB, zG, zBe, outP, N);
}
